// GlobalMeanAttentionPooling_73461120631368
// MI455X (gfx1250) — matrix operand order repaired and checked by reading, not compiled, not run on hardware
//
#include <hip/hip_runtime.h>
#include <hip/hip_bf16.h>

typedef __attribute__((ext_vector_type(16))) __bf16          v16bf;
typedef __attribute__((ext_vector_type(16))) unsigned short  v16u;
typedef __attribute__((ext_vector_type(8)))  float           v8f;

#define NSEG  32
#define DIMS  128
#define CHUNK 32    // rows per WMMA K-step (16x16x32 bf16)
#define NREP  32    // accumulator replicas (atomic-contention relief)

// workspace layout:
//   sums_part  : NREP * NSEG * DIMS floats
//   counts_part: NREP * NSEG        floats
#define SUMS_ELEMS   (NREP * NSEG * DIMS)
#define COUNTS_ELEMS (NREP * NSEG)
#define WS_ELEMS     (SUMS_ELEMS + COUNTS_ELEMS)

// ---------------------------------------------------------------------------
__global__ void gm_zero_ws(float* __restrict__ ws) {
    int i = blockIdx.x * blockDim.x + threadIdx.x;
    if (i < WS_ELEMS) ws[i] = 0.0f;
}

// ---------------------------------------------------------------------------
// Per-segment counts via LDS histogram, flushed into a replicated accumulator.
// ---------------------------------------------------------------------------
__global__ void gm_count(const int* __restrict__ idx, float* __restrict__ counts_part,
                         int n) {
    __shared__ int hist[NSEG];
    if (threadIdx.x < NSEG) hist[threadIdx.x] = 0;
    __syncthreads();
    for (int i = blockIdx.x * blockDim.x + threadIdx.x; i < n;
         i += gridDim.x * blockDim.x) {
        int s = idx[i];
        if (s >= 0 && s < NSEG) atomicAdd(&hist[s], 1);
    }
    __syncthreads();
    if (threadIdx.x < NSEG) {
        int h = hist[threadIdx.x];
        if (h) unsafeAtomicAdd(&counts_part[(blockIdx.x & (NREP - 1)) * NSEG + threadIdx.x],
                               (float)h);
    }
}

// ---------------------------------------------------------------------------
// helper (inlined): split a packed u64 (two f32) into bf16 hi/lo pieces
// ---------------------------------------------------------------------------
struct Split4 { unsigned short hx, lx, hy, ly; };

__device__ __forceinline__ Split4 split2(unsigned long long u) {
    unsigned fbx = (unsigned)u;
    unsigned fby = (unsigned)(u >> 32);
    unsigned hbx = fbx & 0xFFFF0000u;
    unsigned hby = fby & 0xFFFF0000u;
    float lfx = __builtin_bit_cast(float, fbx) - __builtin_bit_cast(float, hbx);
    float lfy = __builtin_bit_cast(float, fby) - __builtin_bit_cast(float, hby);
    Split4 r;
    r.hx = (unsigned short)(hbx >> 16);
    r.hy = (unsigned short)(hby >> 16);
    r.lx = (unsigned short)(__builtin_bit_cast(unsigned, lfx) >> 16);
    r.ly = (unsigned short)(__builtin_bit_cast(unsigned, lfy) >> 16);
    return r;
}

// ---------------------------------------------------------------------------
// Segment-sum as onehot(idx) x features via v_wmma_f32_16x16x32_bf16.
// fp32 features split into bf16 hi + bf16 lo, chained into one f32 accumulator.
//
// Workgroup = 128 threads = 4 waves. Wave w owns dims [32w, 32w+32) with
// interleaved column mapping: lane n loads {dim 2n, 2n+1} as one b64; .x feeds
// dim-tile 0, .y feeds dim-tile 1 (128B per half-wave per load, NT hint).
// Two M-tiles cover segments 0-15 / 16-31 -> 4 accumulators, 8 WMMAs/chunk.
// ---------------------------------------------------------------------------
__global__ void __launch_bounds__(128)
gm_segsum_wmma(const float* __restrict__ feat, const int* __restrict__ idx,
               float* __restrict__ sums_part, int n) {
    const int  lane    = threadIdx.x & 31;       // wave32
    const int  wave    = threadIdx.x >> 5;       // 0..3
    const int  dimbase = wave * 32;
    const bool lo_half = lane < 16;
    const int  m       = lane & 15;
    const int  ncol    = lane & 15;
    const int  khalf   = lo_half ? 0 : 16;       // this half-wave's K offset

    v8f acc00 = {}, acc01 = {};  // segs 0..15  x dim-tiles {even, odd}
    v8f acc10 = {}, acc11 = {};  // segs 16..31 x dim-tiles {even, odd}

    const int nChunks = (n + CHUNK - 1) / CHUNK;
    for (int c = blockIdx.x; c < nChunks; c += gridDim.x) {
        const int base = c * CHUNK;

        int  sIdx[CHUNK];
        v16u bh0 = {}, bl0 = {}, bh1 = {}, bl1 = {};

        if (base + CHUNK <= n) {
            // ================= HOT PATH: full chunk, no guards ===========
            // idx chunk -> SGPRs (merges to s_load_b512 x2)
#pragma unroll
            for (int k = 0; k < CHUNK; ++k) sIdx[k] = idx[base + k];

            // one per-lane base pointer; 16 NT b64 loads at constant offsets
            const float* fb = feat + (size_t)(base + khalf) * DIMS + dimbase + 2 * ncol;
#pragma unroll
            for (int v = 0; v < 8; ++v) {
#pragma unroll
                for (int h = 0; h < 2; ++h) {
                    const int kk = 2 * v + h;   // K within half (const offset)
                    unsigned long long u = __builtin_nontemporal_load(
                        reinterpret_cast<const unsigned long long*>(fb + (size_t)kk * DIMS));
                    Split4 s = split2(u);
                    bh0[kk] = s.hx; bl0[kk] = s.lx;
                    bh1[kk] = s.hy; bl1[kk] = s.ly;
                }
            }
        } else {
            // ================= COLD TAIL: guarded + clamped ==============
#pragma unroll
            for (int k = 0; k < CHUNK; ++k)
                sIdx[k] = (base + k < n) ? idx[base + k] : -1;
#pragma unroll
            for (int v = 0; v < 8; ++v) {
#pragma unroll
                for (int h = 0; h < 2; ++h) {
                    const int kk = 2 * v + h;
                    int r = base + khalf + kk;
                    if (r > n - 1) r = n - 1;   // clamped; onehot==0 there
                    unsigned long long u = *reinterpret_cast<const unsigned long long*>(
                        feat + (size_t)r * DIMS + dimbase + 2 * ncol);
                    Split4 s = split2(u);
                    bh0[kk] = s.hx; bl0[kk] = s.lx;
                    bh1[kk] = s.hy; bl1[kk] = s.ly;
                }
            }
        }

        // ---- one-hot A tiles (16x32, contiguous K order, same as B) -------------------
        // lanes 0-15 : K=2v,2v+1
        // lanes 16-31: K=16+2v,17+2v
        v16u a0u = {}, a1u = {};
#pragma unroll
        for (int v = 0; v < 8; ++v) {
            const int kb = 2 * v;
            const int i0 = lo_half ? sIdx[kb]     : sIdx[kb + 16];
            const int i1 = lo_half ? sIdx[kb + 1] : sIdx[kb + 17];
            a0u[2 * v]     = (i0 == m)      ? (unsigned short)0x3F80 : (unsigned short)0;
            a0u[2 * v + 1] = (i1 == m)      ? (unsigned short)0x3F80 : (unsigned short)0;
            a1u[2 * v]     = (i0 == m + 16) ? (unsigned short)0x3F80 : (unsigned short)0;
            a1u[2 * v + 1] = (i1 == m + 16) ? (unsigned short)0x3F80 : (unsigned short)0;
        }

        const v16bf a0  = __builtin_bit_cast(v16bf, a0u);
        const v16bf a1  = __builtin_bit_cast(v16bf, a1u);
        const v16bf b0h = __builtin_bit_cast(v16bf, bh0);
        const v16bf b0l = __builtin_bit_cast(v16bf, bl0);
        const v16bf b1h = __builtin_bit_cast(v16bf, bh1);
        const v16bf b1l = __builtin_bit_cast(v16bf, bl1);

        acc00 = __builtin_amdgcn_wmma_f32_16x16x32_bf16(false, a0, false, b0h,
                                                        (short)0, acc00, false, false);
        acc00 = __builtin_amdgcn_wmma_f32_16x16x32_bf16(false, a0, false, b0l,
                                                        (short)0, acc00, false, false);
        acc01 = __builtin_amdgcn_wmma_f32_16x16x32_bf16(false, a0, false, b1h,
                                                        (short)0, acc01, false, false);
        acc01 = __builtin_amdgcn_wmma_f32_16x16x32_bf16(false, a0, false, b1l,
                                                        (short)0, acc01, false, false);
        acc10 = __builtin_amdgcn_wmma_f32_16x16x32_bf16(false, a1, false, b0h,
                                                        (short)0, acc10, false, false);
        acc10 = __builtin_amdgcn_wmma_f32_16x16x32_bf16(false, a1, false, b0l,
                                                        (short)0, acc10, false, false);
        acc11 = __builtin_amdgcn_wmma_f32_16x16x32_bf16(false, a1, false, b1h,
                                                        (short)0, acc11, false, false);
        acc11 = __builtin_amdgcn_wmma_f32_16x16x32_bf16(false, a1, false, b1l,
                                                        (short)0, acc11, false, false);
    }

    // ---- flush (C/D layout: VGPR j -> M = j + 8*(lane>=16), N = lane&15) ----
    float* part = sums_part + (size_t)(blockIdx.x & (NREP - 1)) * (NSEG * DIMS);
    const int col0 = dimbase + 2 * ncol;       // dim-tile 0 (.x) column
    const int col1 = col0 + 1;                 // dim-tile 1 (.y) column
#pragma unroll
    for (int j = 0; j < 8; ++j) {
        const int mm = j + (lo_half ? 0 : 8);
        unsafeAtomicAdd(&part[mm * DIMS + col0],        acc00[j]);
        unsafeAtomicAdd(&part[mm * DIMS + col1],        acc01[j]);
        unsafeAtomicAdd(&part[(16 + mm) * DIMS + col0], acc10[j]);
        unsafeAtomicAdd(&part[(16 + mm) * DIMS + col1], acc11[j]);
    }
}

// ---------------------------------------------------------------------------
// Fold replicas: out[s][d] = sigmoid( (sum_r sums[r][s][d]) / (sum_r cnt[r][s]) )
// ---------------------------------------------------------------------------
__global__ void gm_finalize(const float* __restrict__ ws, float* __restrict__ out) {
    int i = blockIdx.x * blockDim.x + threadIdx.x;
    if (i < NSEG * DIMS) {
        const int s = i / DIMS;
        float sum = 0.0f, cnt = 0.0f;
#pragma unroll 4
        for (int r = 0; r < NREP; ++r) {
            sum += ws[r * (NSEG * DIMS) + i];
            cnt += ws[SUMS_ELEMS + r * NSEG + s];
        }
        float mean = sum / cnt;
        out[i] = 1.0f / (1.0f + expf(-mean));
    }
}

// ---------------------------------------------------------------------------
extern "C" void kernel_launch(void* const* d_in, const int* in_sizes, int n_in,
                              void* d_out, int out_size, void* d_ws, size_t ws_size,
                              hipStream_t stream) {
    const float* feat = (const float*)d_in[0];   // [N, 128] fp32
    const int*   idx  = (const int*)d_in[1];     // [N] int
    const int    n    = in_sizes[1];             // N = 2,000,000

    float* sums_part   = (float*)d_ws;                 // NREP * 32 * 128
    float* counts_part = sums_part + SUMS_ELEMS;       // NREP * 32

    gm_zero_ws<<<(WS_ELEMS + 255) / 256, 256, 0, stream>>>((float*)d_ws);
    gm_count<<<1024, 256, 0, stream>>>(idx, counts_part, n);
    gm_segsum_wmma<<<2048, 128, 0, stream>>>(feat, idx, sums_part, n);
    gm_finalize<<<(NSEG * DIMS + 255) / 256, 256, 0, stream>>>((const float*)d_ws,
                                                               (float*)d_out);
}
